// MultiHeadAttention_75969381531696
// MI455X (gfx1250) — matrix operand order repaired and checked by reading, not compiled, not run on hardware
//
#include <hip/hip_runtime.h>
#include <hip/hip_bf16.h>

// MHA forward for MI455X (gfx1250), wave32 + V_WMMA_F32_16X16X32_BF16.
// Pipeline: [cvt weights f32->bf16] -> [QKV proj GEMM, bf16 out, V transposed]
//           -> [flash attention, async-LDS staged K/V, online softmax]
//           -> [out proj, f32]
// Round 4: fix async builtin arg types — it expects (global v4i*, local v4i*,
// imm offset, imm cpol). Probe confirmed the builtin exists on this toolchain.
// Workspace: 4*2MB weights + 3*16MB Q/K/Vt + 16MB att = ~72MB (<= ws_size).

#define DEVINL __device__ __forceinline__

typedef __attribute__((ext_vector_type(16))) __bf16 v16bf;
typedef __attribute__((ext_vector_type(8)))  __bf16 v8bf;
typedef __attribute__((ext_vector_type(8)))  float  v8f;
typedef __attribute__((ext_vector_type(4)))  int    v4i;
typedef int v4i_ __attribute__((vector_size(16)));   // matches builtin param type

constexpr int Bn = 4, Sn = 2048, Dn = 1024, Hn = 16, DHn = 64;
constexpr int Mn = Bn * Sn;   // 8192 tokens
constexpr int KPAD = 72;      // LDS row stride (bf16) for 64-wide tiles

#if __has_builtin(__builtin_amdgcn_global_load_async_to_lds_b128)
#define HAS_ASYNC_LDS 1
#else
#define HAS_ASYNC_LDS 0
#endif

DEVINL v8f wmma_bf16(v16bf a, v16bf b, v8f c) {
  // D = A(16x32) * B(32x16) + C, f32 accumulate
  return __builtin_amdgcn_wmma_f32_16x16x32_bf16(false, a, false, b, (short)0, c,
                                                 false, false);
}

// A-fragment (16x32 bf16): lane holds row m=lane%16; 16 contiguous k at hl*16
DEVINL v16bf a_frag_bf16(const __bf16* rowk, int hl) {
  v8bf lo = *(const v8bf*)(rowk + hl * 16);
  v8bf hi = *(const v8bf*)(rowk + hl * 16 + 8);
  v16bf a;
#pragma unroll
  for (int i = 0; i < 8; ++i) { a[i] = lo[i]; a[i + 8] = hi[i]; }
  return a;
}

// Same, converting from f32 source on the fly
DEVINL v16bf a_frag_f32(const float* rowk, int hl) {
  v8f lo = *(const v8f*)(rowk + hl * 16);
  v8f hi = *(const v8f*)(rowk + hl * 16 + 8);
  v16bf a;
#pragma unroll
  for (int i = 0; i < 8; ++i) { a[i] = (__bf16)lo[i]; a[i + 8] = (__bf16)hi[i]; }
  return a;
}

// B-fragment (32x16 bf16): lane holds col n=lane%16; 16 contiguous k at hl*16.
DEVINL v16bf b_frag(const __bf16* p) { return *(const v16bf*)p; }

// 16B copy global -> LDS (async on CDNA5; fallback goes through VGPRs)
DEVINL void cp16_g2l(__bf16* lds, const __bf16* gsrc) {
#if HAS_ASYNC_LDS
  __builtin_amdgcn_global_load_async_to_lds_b128(
      (__attribute__((address_space(1))) v4i_*)(void*)gsrc,
      (__attribute__((address_space(3))) v4i_*)(void*)lds, 0, 0);
#else
  *(v4i*)lds = *(const v4i*)gsrc;
#endif
}

DEVINL void wait_async_done() {
#if HAS_ASYNC_LDS
#if __has_builtin(__builtin_amdgcn_s_wait_asynccnt)
  __builtin_amdgcn_s_wait_asynccnt(0);
#else
  asm volatile("s_wait_asynccnt 0x0" ::: "memory");
#endif
#endif
}

// Stage one 64x64 bf16 tile (row stride srcstride) into LDS (row stride KPAD).
// 512 x 16B chunks, 2 per thread.
DEVINL void stage_tile(__bf16* dst, const __bf16* src, int srcstride, int tid) {
#pragma unroll
  for (int i = 0; i < 2; ++i) {
    int c = tid + i * 256;
    int row = c >> 3, seg = c & 7;
    cp16_g2l(dst + row * KPAD + seg * 8, src + (size_t)row * srcstride + seg * 8);
  }
}

// ---------------------------------------------------------------- weight cvt
__global__ void cvt_f32_bf16(const float* __restrict__ src, __bf16* __restrict__ dst,
                             int n) {
  int i = blockIdx.x * blockDim.x + threadIdx.x;
  if (i < n) dst[i] = (__bf16)src[i];
}

// --------------------------------------------------------------- QKV proj GEMM
// y = X @ W.T ; X (Mn x Dn) f32, W bf16 (N x K row-major = B^T directly).
// mode 0: out[((b*H+h)*S + s)*64 + dh]  (Q, K as (B,H,S,DH))
// mode 1: out[((b*H+h)*64 + dh)*S + s]  (V transposed: (B,H,DH,S))
// 256 thr = 8 waves; wave computes 32(M)x64(N); WG tile 128 x 128.
__global__ void proj_qkv(const float* __restrict__ X, const __bf16* __restrict__ Wb,
                         __bf16* __restrict__ out, int mode) {
  int tid = threadIdx.x;
  int lane = tid & 31, wave = tid >> 5;
  int hl = lane >> 4, l16 = lane & 15;
  int m0 = blockIdx.x * 128 + (wave >> 1) * 32;  // 4 waves stacked in M
  int n0 = blockIdx.y * 128 + (wave & 1) * 64;   // 2 waves in N

  const float* Arow0 = X + (size_t)(m0 + l16) * Dn;
  const float* Arow1 = Arow0 + (size_t)16 * Dn;
  const __bf16* Brow[4];
#pragma unroll
  for (int ni = 0; ni < 4; ++ni)
    Brow[ni] = Wb + (size_t)(n0 + ni * 16 + l16) * Dn + hl * 16;

  v8f acc[2][4];
#pragma unroll
  for (int mi = 0; mi < 2; ++mi)
#pragma unroll
    for (int ni = 0; ni < 4; ++ni) acc[mi][ni] = 0;

  for (int kk = 0; kk < Dn; kk += 32) {
    v16bf a0 = a_frag_f32(Arow0 + kk, hl);
    v16bf a1 = a_frag_f32(Arow1 + kk, hl);
#pragma unroll
    for (int ni = 0; ni < 4; ++ni) {
      v16bf b = b_frag(Brow[ni] + kk);
      acc[0][ni] = wmma_bf16(a0, b, acc[0][ni]);
      acc[1][ni] = wmma_bf16(a1, b, acc[1][ni]);
    }
  }

  // div-free epilogue: a 32-row tile never crosses a batch boundary (2048|32)
  int bb = m0 >> 11;            // m0 / Sn (uniform)
  int sb = m0 & (Sn - 1);       // m0 % Sn (uniform)
#pragma unroll
  for (int mi = 0; mi < 2; ++mi)
#pragma unroll
    for (int ni = 0; ni < 4; ++ni) {
#pragma unroll
      for (int r = 0; r < 8; ++r) {
        int s = sb + mi * 16 + hl * 8 + r;  // seq position
        int d = n0 + ni * 16 + l16;         // output feature
        int h = d >> 6, dh = d & 63;
        __bf16 v = (__bf16)acc[mi][ni][r];
        if (mode == 0)
          out[(size_t)((bb * Hn + h) * Sn + s) * DHn + dh] = v;
        else
          out[(size_t)((bb * Hn + h) * DHn + dh) * Sn + s] = v;
      }
    }
}

// ------------------------------------------------------------- flash attention
// Grid: (S/256, H, B); 256 thr = 8 waves; wave owns 32 query rows x DH=64.
// All waves run a uniform key loop; K/V 64x64 tiles are staged once per WG in
// LDS (double-buffered, async copies), then fragments are read from LDS.
__global__ void flash_attn(const __bf16* __restrict__ Q, const __bf16* __restrict__ K,
                           const __bf16* __restrict__ Vt, const int* __restrict__ pmask,
                           __bf16* __restrict__ attout) {
  __shared__ __align__(16) __bf16 Ksh[2][64 * KPAD];  // 2 x 9KB
  __shared__ __align__(16) __bf16 Vsh[2][64 * KPAD];  // 2 x 9KB
  __shared__ __align__(16) __bf16 Plds[8][32 * 64];   // 32KB
  int tid = threadIdx.x;
  int lane = tid & 31, wave = tid >> 5;
  int hl = lane >> 4, l16 = lane & 15;
  int b = blockIdx.z, h = blockIdx.y;
  int q0 = blockIdx.x * 256 + wave * 32;

  const __bf16* Qb = Q + (size_t)(b * Hn + h) * Sn * DHn;
  const __bf16* Kb = K + (size_t)(b * Hn + h) * Sn * DHn;
  const __bf16* Vb = Vt + (size_t)(b * Hn + h) * DHn * Sn;
  __bf16* Pw = Plds[wave];

  v16bf qa[2][2];
#pragma unroll
  for (int qt = 0; qt < 2; ++qt) {
    const __bf16* Qrow = Qb + (size_t)(q0 + qt * 16 + l16) * DHn;
    qa[qt][0] = a_frag_bf16(Qrow, hl);       // k = 0..31 of DH
    qa[qt][1] = a_frag_bf16(Qrow + 32, hl);  // k = 32..63
  }

  v8f O[2][4];
  float mrow[2][8], lrow[2][8];
#pragma unroll
  for (int qt = 0; qt < 2; ++qt) {
#pragma unroll
    for (int dt = 0; dt < 4; ++dt) O[qt][dt] = 0;
#pragma unroll
    for (int r = 0; r < 8; ++r) { mrow[qt][r] = -3.0e38f; lrow[qt][r] = 0.0f; }
  }

  int kbmax_own = (q0 + 31) >> 6;      // last block this wave really needs
  int kbmax_wg  = blockIdx.x * 4 + 3;  // uniform WG loop bound

  // prologue: issue async staging of key-block 0 into buffer 0
  stage_tile(Ksh[0], Kb, DHn, tid);
  stage_tile(Vsh[0], Vb, Sn, tid);

  for (int kb = 0; kb <= kbmax_wg; ++kb) {
    int cur = kb & 1;
    int k0 = kb << 6;

    // tile `cur` written by all waves; everyone done reading `cur^1` (iter kb-1)
    wait_async_done();
    __syncthreads();
    if (kb < kbmax_wg) {  // overlap: prefetch next block into the other buffer
      stage_tile(Ksh[cur ^ 1], Kb + (size_t)(k0 + 64) * DHn, DHn, tid);
      stage_tile(Vsh[cur ^ 1], Vb + (k0 + 64), Sn, tid);
    }
    const __bf16* Kt = Ksh[cur];
    const __bf16* Vtl = Vsh[cur];

    // S_ij = Q K^T : 4 key-tiles x 2 q-tiles, K fragments from LDS
    v8f Sc[2][4];
#pragma unroll
    for (int nt = 0; nt < 4; ++nt) {
      const __bf16* Krow = Kt + (nt * 16 + l16) * KPAD;
      v16bf kf0 = b_frag(Krow + hl * 16);
      v16bf kf1 = b_frag(Krow + 32 + hl * 16);
#pragma unroll
      for (int qt = 0; qt < 2; ++qt) {
        v8f c = 0;
        c = wmma_bf16(qa[qt][0], kf0, c);
        c = wmma_bf16(qa[qt][1], kf1, c);
        Sc[qt][nt] = c;
      }
    }

    // padding mask values, shared by both q-tiles
    int pd[4];
#pragma unroll
    for (int nt = 0; nt < 4; ++nt) pd[nt] = pmask[b * Sn + k0 + nt * 16 + l16];

    // scale + mask. Causal compare only needed at/past this wave's diagonal
    // block (also fully masks the uniform-loop overshoot blocks).
    if (kb >= kbmax_own) {
#pragma unroll
      for (int qt = 0; qt < 2; ++qt)
#pragma unroll
        for (int nt = 0; nt < 4; ++nt) {
          int kg = k0 + nt * 16 + l16;
#pragma unroll
          for (int r = 0; r < 8; ++r) {
            int qg = q0 + qt * 16 + hl * 8 + r;
            float s = Sc[qt][nt][r] * 0.125f;  // 1/sqrt(64)
            if (kg > qg || pd[nt] == 0) s = -1.0e9f;
            Sc[qt][nt][r] = s;
          }
        }
    } else {
#pragma unroll
      for (int qt = 0; qt < 2; ++qt)
#pragma unroll
        for (int nt = 0; nt < 4; ++nt) {
          bool ok = (pd[nt] != 0);
#pragma unroll
          for (int r = 0; r < 8; ++r) {
            float s = Sc[qt][nt][r] * 0.125f;
            Sc[qt][nt][r] = ok ? s : -1.0e9f;
          }
        }
    }

    // online softmax per q-tile: row max via half-wave butterfly
#pragma unroll
    for (int qt = 0; qt < 2; ++qt) {
      float alpha[8];
#pragma unroll
      for (int r = 0; r < 8; ++r) {
        float v = fmaxf(fmaxf(Sc[qt][0][r], Sc[qt][1][r]),
                        fmaxf(Sc[qt][2][r], Sc[qt][3][r]));
        v = fmaxf(v, __shfl_xor(v, 1, 32));
        v = fmaxf(v, __shfl_xor(v, 2, 32));
        v = fmaxf(v, __shfl_xor(v, 4, 32));
        v = fmaxf(v, __shfl_xor(v, 8, 32));
        float mnew = fmaxf(mrow[qt][r], v);
        alpha[r] = __expf(mrow[qt][r] - mnew);
        mrow[qt][r] = mnew;
      }
#pragma unroll
      for (int dt = 0; dt < 4; ++dt)
#pragma unroll
        for (int r = 0; r < 8; ++r) O[qt][dt][r] *= alpha[r];

      // P = exp(S - m); row-sum; spill P to LDS (row-major 32x64 bf16)
      float psum[8];
#pragma unroll
      for (int r = 0; r < 8; ++r) psum[r] = 0.0f;
#pragma unroll
      for (int nt = 0; nt < 4; ++nt)
#pragma unroll
        for (int r = 0; r < 8; ++r) {
          float p = __expf(Sc[qt][nt][r] - mrow[qt][r]);
          psum[r] += p;
          Pw[(qt * 16 + hl * 8 + r) * 64 + nt * 16 + l16] = (__bf16)p;
        }
#pragma unroll
      for (int r = 0; r < 8; ++r) {
        float v = psum[r];
        v += __shfl_xor(v, 1, 32);
        v += __shfl_xor(v, 2, 32);
        v += __shfl_xor(v, 4, 32);
        v += __shfl_xor(v, 8, 32);
        lrow[qt][r] = lrow[qt][r] * alpha[r] + v;
      }
    }

    // wave-private LDS turnaround: ensure P stores landed before A-frag reads
    asm volatile("s_wait_dscnt 0" ::: "memory");

    // O += P @ V : A-frags from LDS, V fragments from LDS (shared by q-tiles)
    v16bf pa[2][2];
#pragma unroll
    for (int qt = 0; qt < 2; ++qt) {
      const __bf16* Prow = Pw + (qt * 16 + l16) * 64;
      pa[qt][0] = a_frag_bf16(Prow, hl);
      pa[qt][1] = a_frag_bf16(Prow + 32, hl);
    }
#pragma unroll
    for (int dt = 0; dt < 4; ++dt) {
      const __bf16* Vrow = Vtl + (dt * 16 + l16) * KPAD;
      v16bf vf0 = b_frag(Vrow + hl * 16);
      v16bf vf1 = b_frag(Vrow + 32 + hl * 16);
#pragma unroll
      for (int qt = 0; qt < 2; ++qt) {
        O[qt][dt] = wmma_bf16(pa[qt][0], vf0, O[qt][dt]);
        O[qt][dt] = wmma_bf16(pa[qt][1], vf1, O[qt][dt]);
      }
    }
  }

  // epilogue: normalize and store att as (B,S,H*DH) bf16
#pragma unroll
  for (int qt = 0; qt < 2; ++qt) {
    float inv[8];
#pragma unroll
    for (int r = 0; r < 8; ++r) inv[r] = 1.0f / lrow[qt][r];
#pragma unroll
    for (int dt = 0; dt < 4; ++dt)
#pragma unroll
      for (int r = 0; r < 8; ++r) {
        int qg = q0 + qt * 16 + hl * 8 + r;
        int d = h * DHn + dt * 16 + l16;
        attout[(size_t)(b * Sn + qg) * Dn + d] = (__bf16)(O[qt][dt][r] * inv[r]);
      }
  }
}

// ----------------------------------------------------------- output projection
// out = att @ Wo.T ; att bf16 (Mn x Dn), Wo bf16 (N x K = B^T), out f32.
__global__ void out_proj(const __bf16* __restrict__ Att, const __bf16* __restrict__ Wb,
                         float* __restrict__ out) {
  int tid = threadIdx.x;
  int lane = tid & 31, wave = tid >> 5;
  int hl = lane >> 4, l16 = lane & 15;
  int m0 = blockIdx.x * 128 + (wave >> 1) * 32;
  int n0 = blockIdx.y * 128 + (wave & 1) * 64;

  const __bf16* Arow0 = Att + (size_t)(m0 + l16) * Dn;
  const __bf16* Arow1 = Arow0 + (size_t)16 * Dn;
  const __bf16* Brow[4];
#pragma unroll
  for (int ni = 0; ni < 4; ++ni)
    Brow[ni] = Wb + (size_t)(n0 + ni * 16 + l16) * Dn + hl * 16;

  v8f acc[2][4];
#pragma unroll
  for (int mi = 0; mi < 2; ++mi)
#pragma unroll
    for (int ni = 0; ni < 4; ++ni) acc[mi][ni] = 0;

  for (int kk = 0; kk < Dn; kk += 32) {
    v16bf a0 = a_frag_bf16(Arow0 + kk, hl);
    v16bf a1 = a_frag_bf16(Arow1 + kk, hl);
#pragma unroll
    for (int ni = 0; ni < 4; ++ni) {
      v16bf b = b_frag(Brow[ni] + kk);
      acc[0][ni] = wmma_bf16(a0, b, acc[0][ni]);
      acc[1][ni] = wmma_bf16(a1, b, acc[1][ni]);
    }
  }

#pragma unroll
  for (int mi = 0; mi < 2; ++mi)
#pragma unroll
    for (int ni = 0; ni < 4; ++ni)
#pragma unroll
      for (int r = 0; r < 8; ++r) {
        int m = m0 + mi * 16 + hl * 8 + r;
        int d = n0 + ni * 16 + l16;
        out[(size_t)m * Dn + d] = acc[mi][ni][r];
      }
}

// ------------------------------------------------------------------- launcher
extern "C" void kernel_launch(void* const* d_in, const int* in_sizes, int n_in,
                              void* d_out, int out_size, void* d_ws, size_t ws_size,
                              hipStream_t stream) {
  const float* q_raw  = (const float*)d_in[0];
  const float* kv_raw = (const float*)d_in[1];
  const int*   pmask  = (const int*)d_in[2];
  const float* Wq = (const float*)d_in[3];
  const float* Wk = (const float*)d_in[4];
  const float* Wv = (const float*)d_in[5];
  const float* Wo = (const float*)d_in[6];
  float* out = (float*)d_out;

  char* ws = (char*)d_ws;
  size_t off = 0;
  auto carve = [&](size_t bytes) {
    char* p = ws + off;
    off += (bytes + 255) & ~(size_t)255;
    return p;
  };
  __bf16* Wq_b = (__bf16*)carve((size_t)Dn * Dn * 2);
  __bf16* Wk_b = (__bf16*)carve((size_t)Dn * Dn * 2);
  __bf16* Wv_b = (__bf16*)carve((size_t)Dn * Dn * 2);
  __bf16* Wo_b = (__bf16*)carve((size_t)Dn * Dn * 2);
  __bf16* Qb   = (__bf16*)carve((size_t)Mn * Dn * 2);  // (B,H,S,DH)
  __bf16* Kb   = (__bf16*)carve((size_t)Mn * Dn * 2);  // (B,H,S,DH)
  __bf16* Vtb  = (__bf16*)carve((size_t)Mn * Dn * 2);  // (B,H,DH,S)
  __bf16* Ab   = (__bf16*)carve((size_t)Mn * Dn * 2);  // (B,S,D)

  const int ce = Dn * Dn;  // 1M elems per weight
  cvt_f32_bf16<<<ce / 256, 256, 0, stream>>>(Wq, Wq_b, ce);
  cvt_f32_bf16<<<ce / 256, 256, 0, stream>>>(Wk, Wk_b, ce);
  cvt_f32_bf16<<<ce / 256, 256, 0, stream>>>(Wv, Wv_b, ce);
  cvt_f32_bf16<<<ce / 256, 256, 0, stream>>>(Wo, Wo_b, ce);

  dim3 pgrid(Mn / 128, Dn / 128);
  proj_qkv<<<pgrid, 256, 0, stream>>>(q_raw,  Wq_b, Qb,  0);
  proj_qkv<<<pgrid, 256, 0, stream>>>(kv_raw, Wk_b, Kb,  0);
  proj_qkv<<<pgrid, 256, 0, stream>>>(kv_raw, Wv_b, Vtb, 1);

  flash_attn<<<dim3(Sn / 256, Hn, Bn), 256, 0, stream>>>(Qb, Kb, Vtb, pmask, Ab);

  out_proj<<<pgrid, 256, 0, stream>>>(Ab, Wo_b, out);
}
